// HybridBinaryClassifier_65481071407448
// MI455X (gfx1250) — matrix operand order repaired and checked by reading, not compiled, not run on hardware
//
#include <hip/hip_runtime.h>
#include <hip/hip_bf16.h>

// ---------------------------------------------------------------------------
// Pipeline: conv1(5x5,s2,p1)+relu+pool(2,1) -> conv2(3x3,s2,p1)+relu+pool(2,1)
//           -> FC1 (split-precision f16 WMMA, ~f32 accuracy)
//           -> FC2/FC3 + quantum head -> fidelity-graph aggregation
// ---------------------------------------------------------------------------

typedef _Float16 v16h __attribute__((ext_vector_type(16)));
typedef _Float16 v8h  __attribute__((ext_vector_type(8)));
typedef float    v8f  __attribute__((ext_vector_type(8)));

#define KP 55840          // 55815 padded to multiple of 32
#define KREAL 55815       // 15*61*61

// ---------------- conv1 + relu + maxpool(k2,s1), fused per tile -------------
// x:[256,3,250,250] -> pool1:[256,6,123,123]
__global__ __launch_bounds__(256) void k_conv1pool(
    const float* __restrict__ x, const float* __restrict__ w,
    const float* __restrict__ bias, float* __restrict__ out)
{
    __shared__ float sIn[3][37][37];    // input tile (stride2, k5, pad1)
    __shared__ float sConv[6][17][17];  // conv outputs (pre-pool)
    __shared__ float sW[6 * 3 * 25];

    const int b   = blockIdx.z;
    const int pi0 = blockIdx.y * 16;
    const int pj0 = blockIdx.x * 16;
    const int tid = threadIdx.x;

    for (int i = tid; i < 450; i += 256) sW[i] = w[i];

    for (int idx = tid; idx < 3 * 37 * 37; idx += 256) {
        int c = idx / 1369, rem = idx - c * 1369;
        int r = rem / 37,   q   = rem - r * 37;
        int gr = 2 * pi0 - 1 + r, gq = 2 * pj0 - 1 + q;
        float v = 0.f;
        if ((unsigned)gr < 250u && (unsigned)gq < 250u)
            v = x[(((size_t)b * 3 + c) * 250 + gr) * 250 + gq];
        sIn[c][r][q] = v;
    }
    __syncthreads();

    for (int idx = tid; idx < 6 * 289; idx += 256) {
        int o = idx / 289, rem = idx - o * 289;
        int ci = rem / 17, cj = rem - ci * 17;
        float s = bias[o];
        #pragma unroll
        for (int c = 0; c < 3; ++c)
            #pragma unroll
            for (int ki = 0; ki < 5; ++ki)
                #pragma unroll
                for (int kj = 0; kj < 5; ++kj)
                    s += sIn[c][2 * ci + ki][2 * cj + kj] *
                         sW[((o * 3 + c) * 5 + ki) * 5 + kj];
        sConv[o][ci][cj] = s;
    }
    __syncthreads();

    for (int idx = tid; idx < 6 * 256; idx += 256) {
        int o = idx >> 8, rem = idx & 255;
        int pi = rem >> 4, pj = rem & 15;
        int gi = pi0 + pi, gj = pj0 + pj;
        if (gi < 123 && gj < 123) {
            float m = fmaxf(fmaxf(sConv[o][pi][pj],     sConv[o][pi][pj + 1]),
                            fmaxf(sConv[o][pi + 1][pj], sConv[o][pi + 1][pj + 1]));
            out[(((size_t)b * 6 + o) * 123 + gi) * 123 + gj] = fmaxf(m, 0.f);
        }
    }
}

// ---------------- conv2 + relu + maxpool, writes split-f16 A matrix ---------
// pool1:[256,6,123,123] -> A_hi/A_lo:[256,KP] f16 (flatten o*3721 + i*61 + j)
__global__ __launch_bounds__(256) void k_conv2pool(
    const float* __restrict__ in, const float* __restrict__ w,
    const float* __restrict__ bias,
    _Float16* __restrict__ Ahi, _Float16* __restrict__ Alo)
{
    __shared__ float sIn[6][35][35];
    __shared__ float sConv[15][17][17];
    __shared__ float sW[15 * 6 * 9];

    const int b   = blockIdx.z;
    const int pi0 = blockIdx.y * 16;
    const int pj0 = blockIdx.x * 16;
    const int tid = threadIdx.x;

    for (int i = tid; i < 810; i += 256) sW[i] = w[i];

    for (int idx = tid; idx < 6 * 35 * 35; idx += 256) {
        int c = idx / 1225, rem = idx - c * 1225;
        int r = rem / 35,   q   = rem - r * 35;
        int gr = 2 * pi0 - 1 + r, gq = 2 * pj0 - 1 + q;
        float v = 0.f;
        if ((unsigned)gr < 123u && (unsigned)gq < 123u)
            v = in[(((size_t)b * 6 + c) * 123 + gr) * 123 + gq];
        sIn[c][r][q] = v;
    }
    __syncthreads();

    for (int idx = tid; idx < 15 * 289; idx += 256) {
        int o = idx / 289, rem = idx - o * 289;
        int ci = rem / 17, cj = rem - ci * 17;
        float s = bias[o];
        #pragma unroll
        for (int c = 0; c < 6; ++c)
            #pragma unroll
            for (int ki = 0; ki < 3; ++ki)
                #pragma unroll
                for (int kj = 0; kj < 3; ++kj)
                    s += sIn[c][2 * ci + ki][2 * cj + kj] *
                         sW[((o * 6 + c) * 3 + ki) * 3 + kj];
        sConv[o][ci][cj] = s;
    }
    __syncthreads();

    for (int idx = tid; idx < 15 * 256; idx += 256) {
        int o = idx >> 8, rem = idx & 255;
        int pi = rem >> 4, pj = rem & 15;
        int gi = pi0 + pi, gj = pj0 + pj;
        if (gi < 61 && gj < 61) {
            float m = fmaxf(fmaxf(sConv[o][pi][pj],     sConv[o][pi][pj + 1]),
                            fmaxf(sConv[o][pi + 1][pj], sConv[o][pi + 1][pj + 1]));
            float v = fmaxf(m, 0.f);
            _Float16 hi = (_Float16)v;
            _Float16 lo = (_Float16)(v - (float)hi);   // Dekker split residual
            size_t k = (size_t)b * KP + (size_t)o * 3721 + (size_t)gi * 61 + gj;
            Ahi[k] = hi;
            Alo[k] = lo;
        }
    }
}

// ---------------- weight conversion (split) / padding -----------------------
__global__ __launch_bounds__(256) void k_cvt_w1(const float* __restrict__ w,
                                                _Float16* __restrict__ Whi,
                                                _Float16* __restrict__ Wlo)
{
    size_t idx = (size_t)blockIdx.x * 256 + threadIdx.x;
    if (idx >= (size_t)128 * KP) return;
    int n = (int)(idx / KP), k = (int)(idx % KP);
    float v = (n < 120 && k < KREAL) ? w[(size_t)n * KREAL + k] : 0.f;
    _Float16 hi = (_Float16)v;
    Whi[idx] = hi;
    Wlo[idx] = (_Float16)(v - (float)hi);
}

__global__ __launch_bounds__(256) void k_padA(_Float16* __restrict__ Ahi,
                                              _Float16* __restrict__ Alo)
{
    int idx = blockIdx.x * 256 + threadIdx.x;
    if (idx >= 256 * (KP - KREAL)) return;
    int b = idx / (KP - KREAL), k = KREAL + idx % (KP - KREAL);
    Ahi[(size_t)b * KP + k] = (_Float16)0.f;
    Alo[(size_t)b * KP + k] = (_Float16)0.f;
}

// ---------------- FC1: split-precision WMMA GEMM ----------------------------
// D = Ahi*Bhi + Ahi*Blo + Alo*Bhi, each term in its OWN f32 accumulator so
// every v_wmma writes a distinct register block exactly once per K-step
// (no D=C rename pressure -> no v_mov/v_nop hazard padding).
// Each wave: 16x16 tile, 3 WMMAs per K-step; 128 waves (16 Mt x 8 Nt).
static __device__ __forceinline__ v16h load_afrag(const _Float16* row, int k, int aoff)
{
    v8h lo = *(const v8h*)(row + k + aoff);
    v8h hi = *(const v8h*)(row + k + 16 + aoff);
    v16h r;
    #pragma unroll
    for (int i = 0; i < 8; ++i) { r[i] = lo[i]; r[8 + i] = hi[i]; }
    return r;
}

static __device__ __forceinline__ v16h load_bfrag(const _Float16* row, int k, int woff)
{
    v8h lo = *(const v8h*)(row + k + woff);
    v8h hi = *(const v8h*)(row + k + woff + 16);
    v16h r;
    #pragma unroll
    for (int i = 0; i < 8; ++i) { r[i] = lo[i]; r[8 + i] = hi[i]; }
    return r;
}

#define WMMA_F16(A_, B_, C_) \
    __builtin_amdgcn_wmma_f32_16x16x32_f16(false, (A_), false, (B_), (short)0, (C_), false, false)

__global__ __launch_bounds__(128) void k_fc1_wmma(
    const _Float16* __restrict__ Ahi, const _Float16* __restrict__ Alo,
    const _Float16* __restrict__ Whi, const _Float16* __restrict__ Wlo,
    const float* __restrict__ bias, float* __restrict__ out)
{
    const int lane  = threadIdx.x & 31;
    const int wv    = threadIdx.x >> 5;
    const int gw    = blockIdx.x * 4 + wv;   // 0..127
    const int mt    = (gw >> 3) * 16;        // 0..240
    const int nt    = (gw & 7) * 16;         // 0..112
    const int half  = lane >> 4;             // 0/1
    const int idx16 = lane & 15;

    const size_t ar = (size_t)(mt + idx16) * KP;
    const size_t wr = (size_t)(nt + idx16) * KP;
    const _Float16* ah_p = Ahi + ar; const _Float16* al_p = Alo + ar;
    const _Float16* bh_p = Whi + wr; const _Float16* bl_p = Wlo + wr;
    const int aoff = half * 8;   // A frag: K = k+aoff..+7 and k+16+aoff..+7
    const int woff = half * 8;   // B frag: K = k+woff..+7 and k+16+woff..+7

    v8f accA = {};   // ah*bh
    v8f accB = {};   // al*bh
    v8f accC = {};   // ah*bl

    for (int k = 0; k < KP; k += 32) {
        __builtin_prefetch(ah_p + k + 512, 0, 1);
        __builtin_prefetch(al_p + k + 512, 0, 1);
        __builtin_prefetch(bh_p + k + 512, 0, 1);
        __builtin_prefetch(bl_p + k + 512, 0, 1);

        v16h ah = load_afrag(ah_p, k, aoff);
        v16h al = load_afrag(al_p, k, aoff);
        v16h bh = load_bfrag(bh_p, k, woff);
        v16h bl = load_bfrag(bl_p, k, woff);

        accB = WMMA_F16(al, bh, accB);
        accC = WMMA_F16(ah, bl, accC);
        accA = WMMA_F16(ah, bh, accA);
    }

    // D layout: element r of lane L -> D[(L>>4)*8 + r][L&15]
    const int n = nt + idx16;
    if (n < 120) {
        float bn = bias[n];
        #pragma unroll
        for (int r = 0; r < 8; ++r) {
            int m = mt + half * 8 + r;
            float v = (accB[r] + accC[r]) + accA[r] + bn;
            out[(size_t)m * 120 + n] = v > 0.f ? v : 0.f;
        }
    }
}

// ---------------- FC2 + FC3 + quantum head ----------------------------------
__global__ __launch_bounds__(128) void k_head(
    const float* __restrict__ fc1o, const float* __restrict__ w2,
    const float* __restrict__ b2,   const float* __restrict__ w3,
    const float* __restrict__ b3,   float* __restrict__ feat,
    float* __restrict__ probs, float* __restrict__ norms)
{
    const int b = blockIdx.x, tid = threadIdx.x;
    __shared__ float s1[120];
    __shared__ float sf[84];
    for (int i = tid; i < 120; i += 128) s1[i] = fc1o[(size_t)b * 120 + i];
    __syncthreads();
    if (tid < 84) {
        float s = b2[tid];
        #pragma unroll 4
        for (int k = 0; k < 120; ++k) s += s1[k] * w2[tid * 120 + k];
        sf[tid] = s;
        feat[(size_t)b * 84 + tid] = s;
    }
    __syncthreads();
    if (tid == 0) {
        float logit = b3[0], nn = 0.f;
        for (int n = 0; n < 84; ++n) { logit += sf[n] * w3[n]; nn += sf[n] * sf[n]; }
        // probs = sigmoid(-sin(logit)) = 1/(1+exp(sin(logit)))
        probs[b] = 1.f / (1.f + expf(sinf(logit)));
        norms[b] = sqrtf(nn);
    }
}

// ---------------- fidelity-graph aggregation --------------------------------
__global__ __launch_bounds__(256) void k_graph(
    const float* __restrict__ feat, const float* __restrict__ probs,
    const float* __restrict__ norms, float* __restrict__ out)
{
    const int i = blockIdx.x, tid = threadIdx.x;
    __shared__ float fi[84];
    __shared__ float rdeg[256];
    __shared__ float rsum[256];
    for (int k = tid; k < 84; k += 256) fi[k] = feat[(size_t)i * 84 + k];
    __syncthreads();

    float d = 0.f, s = 0.f;
    if (tid != i) {
        float dot = 0.f;
        #pragma unroll 4
        for (int k = 0; k < 84; ++k) dot += fi[k] * feat[(size_t)tid * 84 + k];
        float sim = dot / (norms[i] * norms[tid] + 1e-12f);
        if (sim >= 0.9f) { d = 1.f; s = probs[tid]; }
    }
    rdeg[tid] = d; rsum[tid] = s;
    __syncthreads();
    for (int off = 128; off > 0; off >>= 1) {
        if (tid < off) { rdeg[tid] += rdeg[tid + off]; rsum[tid] += rsum[tid + off]; }
        __syncthreads();
    }
    if (tid == 0) {
        float deg = rdeg[0];
        float nm  = deg > 0.f ? rsum[0] / deg : 0.f;
        float agg = (probs[i] + nm) / (1.f + deg);
        out[i * 2 + 0] = agg;
        out[i * 2 + 1] = 1.f - agg;
    }
}

// ---------------------------------------------------------------------------
extern "C" void kernel_launch(void* const* d_in, const int* in_sizes, int n_in,
                              void* d_out, int out_size, void* d_ws, size_t ws_size,
                              hipStream_t stream)
{
    (void)in_sizes; (void)n_in; (void)out_size; (void)ws_size;
    const float* x       = (const float*)d_in[0];
    const float* conv1_w = (const float*)d_in[1];
    const float* conv1_b = (const float*)d_in[2];
    const float* conv2_w = (const float*)d_in[3];
    const float* conv2_b = (const float*)d_in[4];
    const float* fc1_w   = (const float*)d_in[5];
    const float* fc1_b   = (const float*)d_in[6];
    const float* fc2_w   = (const float*)d_in[7];
    const float* fc2_b   = (const float*)d_in[8];
    const float* fc3_w   = (const float*)d_in[9];
    const float* fc3_b   = (const float*)d_in[10];

    // workspace layout (bytes); peak use ~179 MB
    char* ws = (char*)d_ws;
    _Float16* Ahi   = (_Float16*)(ws + 0);               // 256*KP*2 = 28,590,080
    _Float16* Alo   = (_Float16*)(ws + 28590080);        // 28,590,080
    _Float16* Whi   = (_Float16*)(ws + 57180160);        // 128*KP*2 = 14,295,040
    _Float16* Wlo   = (_Float16*)(ws + 71475200);        // 14,295,040
    float*    pool1 = (float*)(ws + 85770240);           // 256*6*123*123*4 = 92,952,576
    float*    fc1o  = (float*)(ws + 178722816);          // 256*120*4 = 122,880
    float*    feat  = (float*)(ws + 178845696);          // 256*84*4  = 86,016
    float*    probs = (float*)(ws + 178931712);          // 1024
    float*    norms = (float*)(ws + 178932736);          // 1024

    k_conv1pool<<<dim3(8, 8, 256), 256, 0, stream>>>(x, conv1_w, conv1_b, pool1);
    k_cvt_w1<<<(128 * KP) / 256, 256, 0, stream>>>(fc1_w, Whi, Wlo);
    k_padA<<<(256 * (KP - KREAL) + 255) / 256, 256, 0, stream>>>(Ahi, Alo);
    k_conv2pool<<<dim3(4, 4, 256), 256, 0, stream>>>(pool1, conv2_w, conv2_b, Ahi, Alo);
    k_fc1_wmma<<<32, 128, 0, stream>>>(Ahi, Alo, Whi, Wlo, fc1_b, fc1o);
    k_head<<<256, 128, 0, stream>>>(fc1o, fc2_w, fc2_b, fc3_w, fc3_b, feat, probs, norms);
    k_graph<<<256, 256, 0, stream>>>(feat, probs, norms, (float*)d_out);
}
